// Predictor_83279415869721
// MI455X (gfx1250) — matrix operand order repaired and checked by reading, not compiled, not run on hardware
//
#include <hip/hip_runtime.h>
#include <hip/hip_bf16.h>

#define HID 512
#define BATCH 4096
#define TSTEPS 20
#define BSTR (HID + 8)   // bf16 LDS row stride (shorts): 1040 B, 16B-aligned, bank-spread

typedef __attribute__((ext_vector_type(16))) __bf16 v16bf;
typedef __attribute__((ext_vector_type(8)))  float  v8f;

union FragU {
    v16bf v;
    unsigned int u[8];
};

__device__ __forceinline__ unsigned short f2bf_rne(float f) {
    unsigned int x = __builtin_bit_cast(unsigned int, f);
    x += 0x7FFFu + ((x >> 16) & 1u);          // round-to-nearest-even
    return (unsigned short)(x >> 16);
}
__device__ __forceinline__ float bf2f(unsigned short h) {
    unsigned int x = ((unsigned int)h) << 16;
    return __builtin_bit_cast(float, x);
}
__device__ __forceinline__ float frcp(float x) {   // fast hw reciprocal (v_rcp_f32)
    return __builtin_amdgcn_rcpf(x);
}
__device__ __forceinline__ float sigmf(float x) {
    return frcp(1.0f + __expf(-x));
}
__device__ __forceinline__ float tanh_fast(float x) {
    float cx = fminf(fmaxf(x, -15.0f), 15.0f);
    float e = __expf(2.0f * cx);
    return (e - 1.0f) * frcp(e + 1.0f);
}

__device__ __forceinline__ v8f wmma_bf16(const FragU& a, const FragU& b, v8f c) {
    // D = A(16x32 bf16) * B(32x16 bf16) + C(16x16 f32)
    return __builtin_amdgcn_wmma_f32_16x16x32_bf16(
        /*neg_a=*/false, a.v, /*neg_b=*/false, b.v,
        /*c_mod=*/(short)0, c, /*reuse_a=*/false, /*reuse_b=*/false);
}

// ---------------- weight fp32 -> bf16 pre-conversion (once per launch) ------
__global__ void convert_weights_kernel(const float* __restrict__ Wi,
                                       const float* __restrict__ Wg,
                                       const float* __restrict__ Wo,
                                       const float* __restrict__ Wout,
                                       unsigned short* __restrict__ wbf) {
    int i = blockIdx.x * blockDim.x + threadIdx.x;
    if (i < HID * HID) {
        wbf[i]                 = f2bf_rne(Wi[i]);
        wbf[i +     HID * HID] = f2bf_rne(Wg[i]);
        wbf[i + 2 * HID * HID] = f2bf_rne(Wo[i]);
        wbf[i + 3 * HID * HID] = f2bf_rne(Wout[i]);
    }
}

// ---------------- main scan: one wave32 owns 16 batch rows for all steps ----
__launch_bounds__(32, 1)
__global__ void node_scan_kernel(const float* __restrict__ y,
                                 const float* __restrict__ ts,
                                 const unsigned short* __restrict__ wbf,
                                 const float* __restrict__ bout,
                                 const float* __restrict__ Wfc,
                                 const float* __restrict__ bfc,
                                 float* __restrict__ out) {
    __shared__ float          zsm[16][HID + 4];  // fp32 ODE state (master copy)
    // `buf` is time-shared: pass1 reads it as the bf16 z-shadow (GEMM1 A),
    // pass2 writes softmax logits u into it, pass3 reads u and rewrites the
    // bf16 z-shadow element-wise strictly after the last read of each element.
    __shared__ unsigned short buf[16][BSTR];
    __shared__ unsigned short dhs[16][BSTR];     // bf16 dh activations (GEMM2 A)

    const int lane = threadIdx.x;
    const int m0   = blockIdx.x * 16;

    const int mrow = lane & 15;
    const int aHi  = (lane >> 4) & 1;     // lane half selects K sub-range
    const int bKhi = aHi ? 16 : 0;
    const int ncol = lane & 15;           // C/D column = lane & 15

    const unsigned short* Wi_b = wbf;
    const unsigned short* Wg_b = wbf +     HID * HID;
    const unsigned short* Wo_b = wbf + 2 * HID * HID;
    const unsigned short* Wu_b = wbf + 3 * HID * HID;

    // ---- load initial z = y into LDS (fp32 master + bf16 shadow) ----
    for (int i = lane; i < 16 * HID; i += 32) {
        int r = i >> 9, c = i & (HID - 1);
        float v = y[(m0 + r) * HID + c];
        zsm[r][c] = v;
        buf[r][c] = f2bf_rne(v);
    }
    __syncthreads();

    const float bfc0 = bfc[0];

    // ---- output head: out[b,t] = softmax(z_t[b]) . Wfc + bfc ----
    auto emit = [&](int t) {
        for (int r = 0; r < 16; ++r) {
            float mx = -1e30f;
            for (int c = lane; c < HID; c += 32) mx = fmaxf(mx, zsm[r][c]);
            #pragma unroll
            for (int off = 16; off; off >>= 1) mx = fmaxf(mx, __shfl_xor(mx, off, 32));
            float se = 0.f, sd = 0.f;
            for (int c = lane; c < HID; c += 32) {
                float e = __expf(zsm[r][c] - mx);
                se += e;
                sd += e * Wfc[c];
            }
            #pragma unroll
            for (int off = 16; off; off >>= 1) {
                se += __shfl_xor(se, off, 32);
                sd += __shfl_xor(sd, off, 32);
            }
            if (lane == 0) out[(m0 + r) * TSTEPS + t] = sd * frcp(se) + bfc0;
        }
    };

    // A fragment (16x32 bf16) from a bf16 LDS buffer: two 16B ds_load_b128.
    // ISA layout: VGPR v<4 hold k = kb..kb+7, v>=4 hold k = kb+16..kb+23,
    // kb = k0 + 8*laneHi -> both groups are 8 contiguous shorts.
    auto loadA = [&](const unsigned short* row, int k0) {
        FragU a;
        const int kb = k0 + (aHi ? 8 : 0);
        uint4 lo = *(const uint4*)(row + kb);
        uint4 hi = *(const uint4*)(row + kb + 16);
        a.u[0] = lo.x; a.u[1] = lo.y; a.u[2] = lo.z; a.u[3] = lo.w;
        a.u[4] = hi.x; a.u[5] = hi.y; a.u[6] = hi.z; a.u[7] = hi.w;
        return a;
    };
    // B fragment (32x16 bf16): k = kb..kb+7 and kb+16..kb+23 (kb = k0 + 8*laneHi) = two global_load_b128.
    auto loadB = [&](const unsigned short* W, int n, int k0) {
        FragU b;
        const unsigned short* p = W + n * HID + k0 + (aHi ? 8 : 0);
        uint4 lo = *(const uint4*)p;
        uint4 hi = *(const uint4*)(p + 16);
        b.u[0] = lo.x; b.u[1] = lo.y; b.u[2] = lo.z; b.u[3] = lo.w;
        b.u[4] = hi.x; b.u[5] = hi.y; b.u[6] = hi.z; b.u[7] = hi.w;
        return b;
    };

    emit(0);  // t = 0 uses the initial state

    #pragma unroll 1
    for (int t = 0; t < TSTEPS - 1; ++t) {
        const float dt = ts[t + 1] - ts[t];

        // ---- pass 1: i/g/o gates + dh, tiled 16(M) x 32(N-pair) x 512(K) ----
        #pragma unroll 1
        for (int ncp = 0; ncp < 16; ++ncp) {
            const int nA = ncp * 32, nB = nA + 16;
            const int nlA = nA + ncol, nlB = nB + ncol;
            v8f ai0 = {}, ai1 = {}, ag0 = {}, ag1 = {}, ao0 = {}, ao1 = {};

            if (ncp < 15) {  // pull next weight chunk toward L2/L0
                __builtin_prefetch(Wi_b + (nA + 32) * HID, 0, 1);
                __builtin_prefetch(Wg_b + (nA + 32) * HID, 0, 1);
                __builtin_prefetch(Wo_b + (nA + 32) * HID, 0, 1);
            }

            for (int kc = 0; kc < HID; kc += 32) {
                FragU a   = loadA(&buf[mrow][0], kc);
                FragU bi0 = loadB(Wi_b, nlA, kc), bi1 = loadB(Wi_b, nlB, kc);
                FragU bg0 = loadB(Wg_b, nlA, kc), bg1 = loadB(Wg_b, nlB, kc);
                FragU bo0 = loadB(Wo_b, nlA, kc), bo1 = loadB(Wo_b, nlB, kc);
                ai0 = wmma_bf16(a, bi0, ai0);
                ai1 = wmma_bf16(a, bi1, ai1);
                ag0 = wmma_bf16(a, bg0, ag0);
                ag1 = wmma_bf16(a, bg1, ag1);
                ao0 = wmma_bf16(a, bo0, ao0);
                ao1 = wmma_bf16(a, bo1, ao1);
            }
            #pragma unroll
            for (int r = 0; r < 8; ++r) {
                int m = r + aHi * 8;
                float i0 = sigmf(ai0[r]), g0 = tanh_fast(ag0[r]), o0 = sigmf(ao0[r]);
                float i1 = sigmf(ai1[r]), g1 = tanh_fast(ag1[r]), o1 = sigmf(ao1[r]);
                dhs[m][nlA] = f2bf_rne(o0 * tanh_fast(i0 * g0));
                dhs[m][nlB] = f2bf_rne(o1 * tanh_fast(i1 * g1));
            }
        }
        __syncthreads();

        // ---- pass 2: u = dh @ Wout.T + bout  (logits written into buf) ----
        #pragma unroll 1
        for (int ncp = 0; ncp < 16; ++ncp) {
            const int nA = ncp * 32, nB = nA + 16;
            const int nlA = nA + ncol, nlB = nB + ncol;
            v8f au0 = {}, au1 = {};
            for (int kc = 0; kc < HID; kc += 32) {
                FragU a  = loadA(&dhs[mrow][0], kc);
                FragU b0 = loadB(Wu_b, nlA, kc);
                FragU b1 = loadB(Wu_b, nlB, kc);
                au0 = wmma_bf16(a, b0, au0);
                au1 = wmma_bf16(a, b1, au1);
            }
            float bo0 = bout[nlA], bo1 = bout[nlB];
            #pragma unroll
            for (int r = 0; r < 8; ++r) {
                int m = r + aHi * 8;
                buf[m][nlA] = f2bf_rne(au0[r] + bo0);
                buf[m][nlB] = f2bf_rne(au1[r] + bo1);
            }
        }
        __syncthreads();

        // ---- pass 3: row softmax of u; z += dt*softmax(u); refresh shadow --
        for (int r = 0; r < 16; ++r) {
            float mx = -1e30f;
            for (int c = lane; c < HID; c += 32) mx = fmaxf(mx, bf2f(buf[r][c]));
            #pragma unroll
            for (int off = 16; off; off >>= 1) mx = fmaxf(mx, __shfl_xor(mx, off, 32));
            float se = 0.f;
            for (int c = lane; c < HID; c += 32) se += __expf(bf2f(buf[r][c]) - mx);
            #pragma unroll
            for (int off = 16; off; off >>= 1) se += __shfl_xor(se, off, 32);
            float scale = dt * frcp(se);
            for (int c = lane; c < HID; c += 32) {
                float u  = bf2f(buf[r][c]);               // last read of element
                float zn = zsm[r][c] + scale * __expf(u - mx);
                zsm[r][c] = zn;
                buf[r][c] = f2bf_rne(zn);                 // becomes z-shadow
            }
        }
        __syncthreads();

        emit(t + 1);
    }
}

extern "C" void kernel_launch(void* const* d_in, const int* in_sizes, int n_in,
                              void* d_out, int out_size, void* d_ws, size_t ws_size,
                              hipStream_t stream) {
    (void)in_sizes; (void)n_in; (void)out_size; (void)ws_size;
    const float* y    = (const float*)d_in[0];
    const float* ts   = (const float*)d_in[1];
    const float* Wi   = (const float*)d_in[2];
    /* d_in[3] = Wf : computed in the reference but unused -> skipped */
    const float* Wg   = (const float*)d_in[4];
    const float* Wo   = (const float*)d_in[5];
    const float* Wout = (const float*)d_in[6];
    const float* bout = (const float*)d_in[7];
    const float* Wfc  = (const float*)d_in[8];
    const float* bfc  = (const float*)d_in[9];

    unsigned short* wbf = (unsigned short*)d_ws;   // 4 * 512*512 bf16 = 2 MB

    convert_weights_kernel<<<(HID * HID + 255) / 256, 256, 0, stream>>>(
        Wi, Wg, Wo, Wout, wbf);

    node_scan_kernel<<<BATCH / 16, 32, 0, stream>>>(
        y, ts, wbf, bout, Wfc, bfc, (float*)d_out);
}
